// MultiRes_26680336843184
// MI455X (gfx1250) — matrix operand order repaired and checked by reading, not compiled, not run on hardware
//
#include <hip/hip_runtime.h>
#include <hip/hip_bf16.h>

// ---------- vector types ----------
typedef __attribute__((ext_vector_type(16))) __bf16 bf16x16;
typedef __attribute__((ext_vector_type(8)))  __bf16 bf16x8;
typedef __attribute__((ext_vector_type(4)))  __bf16 bf16x4;
typedef __attribute__((ext_vector_type(8)))  float  f32x8;
typedef __attribute__((ext_vector_type(4)))  float  f32x4;

// Problem constants: B=2, C=8, F=1024, T=256, Ch=32, Fh=4096
#define T_DIM 256

// ============================================================================
// Kernel 1: LayerNorm over F (affine per (c,f)), writing bf16 activations in
// the WMMA-B packed layout:  h0p[bc][kb = f/32][t][kin = f%32].
// ============================================================================
__global__ __launch_bounds__(256) void layernorm_pack_bf16(
    const float* __restrict__ x,      // (B, C, F, T)
    const float* __restrict__ ln_w,   // (C, F)
    const float* __restrict__ ln_b,   // (C, F)
    __bf16* __restrict__ h0p)         // (B*C, F/32, T, 32) bf16
{
    constexpr int F = 1024, C = 8, KB = F / 32;
    const int bc = blockIdx.x;
    const int c  = bc % C;
    const int t  = threadIdx.x;

    const float* xp = x + (size_t)bc * F * T_DIM + t;
    float s = 0.f, ss = 0.f;
    for (int f = 0; f < F; ++f) {
        float v = xp[(size_t)f * T_DIM];
        s += v; ss += v * v;
    }
    const float mu  = s * (1.0f / F);
    const float var = ss * (1.0f / F) - mu * mu;
    const float r   = rsqrtf(var + 1e-8f);
    const int cF    = c * F;

    for (int kb = 0; kb < KB; ++kb) {
        bf16x8 o[4];
#pragma unroll
        for (int j = 0; j < 32; ++j) {
            const int f = kb * 32 + j;
            float v = (xp[(size_t)f * T_DIM] - mu) * r;
            v = v * ln_w[cF + f] + ln_b[cF + f];
            o[j >> 3][j & 7] = (__bf16)v;
        }
        __bf16* q = h0p + ((size_t)(bc * KB + kb) * T_DIM + t) * 32;
#pragma unroll
        for (int i = 0; i < 4; ++i) *(bf16x8*)&q[i * 8] = o[i];
    }
}

// ============================================================================
// Kernel 2/4: per-channel positionwise GEMM, bf16 WMMA, packed-B activations.
//   Out[bc, g, t] = sum_k W[c, g, k] * Act[bc, k, t]
// Block: 256 threads = 8 waves; block tile 64(g) x 256(t) -> each weight
// element streamed from HBM exactly once; activations re-hit L2 (2MB/bc).
// Wave tile 32x64 = 8 WMMA tiles per K-step (2 A frags x 4 B frags),
// all fragments fetched into registers before the WMMA burst so the
// compiler can issue partial s_wait_dscnt and overlap LDS with WMMA.
// LDS double buffer: As[2][64x32] bf16 (f32 weights cvt in flight),
//                    Bs[2][256x32] bf16 (contiguous packed slab copy).
// Fragment packing follows CDNA5 WMMA VGPR layouts (ISA 7.12.2).
// ============================================================================
template <typename OutT>
__global__ __launch_bounds__(256) void gemm_pw_wmma(
    const float* __restrict__ W,     // (Cn, G, K) f32
    const __bf16* __restrict__ Actp, // (B*Cn, K/32, 256, 32) bf16 packed
    OutT* __restrict__ Out,          // (B*Cn, G, 256)
    int Cn, int G, int K)
{
    constexpr int SLAB = T_DIM * 32;              // 8192 elems = 16KB
    __shared__ __align__(16) __bf16 As[2][64 * 32];
    __shared__ __align__(16) __bf16 Bs[2][SLAB];

    const int tid  = threadIdx.x;
    const int wave = tid >> 5;
    const int lane = tid & 31;
    const int l    = lane & 15;                   // N / M low index
    const int h    = lane >> 4;                   // lane half (K-half select)
    const int wg   = wave & 1;                    // wave g-half (32 rows each)
    const int tw   = wave >> 1;                   // wave t-group (0..3), 64t each

    const int g0 = blockIdx.x * 64;
    const int bc = blockIdx.z;
    const int c  = bc % Cn;

    const float*  Wc = W    + (size_t)c  * G * K;
    const __bf16* Ap = Actp + (size_t)bc * (K / 32) * SLAB;

    // cooperative A-load mapping: 64 rows x 32 k f32 -> 8 f32 per thread
    const int ar = tid >> 2;                      // 0..63 : g row
    const int ak = (tid & 3) * 8;                 // 0/8/16/24 : k offset

    f32x8 acc[2][4] = {{{}, {}, {}, {}}, {{}, {}, {}, {}}};
    const int nk = K / 32;

    // ---- prologue: stage tile kb=0 ----
    {
        const float* wp = Wc + (size_t)(g0 + ar) * K + ak;
        f32x4 w0 = *(const f32x4*)wp;
        f32x4 w1 = *(const f32x4*)(wp + 4);
        bf16x8 a8;
        a8[0] = (__bf16)w0[0]; a8[1] = (__bf16)w0[1];
        a8[2] = (__bf16)w0[2]; a8[3] = (__bf16)w0[3];
        a8[4] = (__bf16)w1[0]; a8[5] = (__bf16)w1[1];
        a8[6] = (__bf16)w1[2]; a8[7] = (__bf16)w1[3];
        *(bf16x8*)&As[0][ar * 32 + ak] = a8;
#pragma unroll
        for (int i = 0; i < 4; ++i) {
            const int off = i * 2048 + tid * 8;
            *(bf16x8*)&Bs[0][off] = *(const bf16x8*)(Ap + off);
        }
    }
    __syncthreads();

    for (int kb = 0; kb < nk; ++kb) {
        const int  cur  = kb & 1;
        const int  nxt  = cur ^ 1;
        const bool more = (kb + 1) < nk;

        // ---- issue global loads for tile kb+1 (latency hidden by WMMAs) ----
        f32x4  wv0 = {}, wv1 = {};
        bf16x8 bv[4];
        if (more) {
            const float* wp = Wc + (size_t)(g0 + ar) * K + (kb + 1) * 32 + ak;
            wv0 = *(const f32x4*)wp;
            wv1 = *(const f32x4*)(wp + 4);
            __builtin_prefetch(wp + 64, 0, 0);    // K tile kb+3 (two past the one loaded)
            const __bf16* sl = Ap + (size_t)(kb + 1) * SLAB;
#pragma unroll
            for (int i = 0; i < 4; ++i)
                bv[i] = *(const bf16x8*)(sl + i * 2048 + tid * 8);
        }

        // ---- fetch ALL fragments for this K-step into registers ----
        // A frag (16x32): lane<16 K=0..7 & 16..23, lane>=16 K=8..15 & 24..31
        bf16x16 af[2];
#pragma unroll
        for (int m = 0; m < 2; ++m) {
            const int arow = (wg * 32 + m * 16 + l) * 32;
            bf16x8 alo = *(const bf16x8*)&As[cur][arow + h * 8];
            bf16x8 ahi = *(const bf16x8*)&As[cur][arow + 16 + h * 8];
            af[m] = __builtin_shufflevector(alo, ahi,
                0,1,2,3,4,5,6,7,8,9,10,11,12,13,14,15);
        }
        // B frag (32x16): lane<16 K=0..7 & 16..23, lane>=16 K=8..15 & 24..31 (N = l)
        bf16x16 bf[4];
#pragma unroll
        for (int n = 0; n < 4; ++n) {
            const int brow = (tw * 64 + n * 16 + l) * 32;
            bf16x8 b0 = *(const bf16x8*)&Bs[cur][brow + h * 8];
            bf16x8 b1 = *(const bf16x8*)&Bs[cur][brow + 16 + h * 8];
            bf[n] = __builtin_shufflevector(b0, b1,
                0,1,2,3,4,5,6,7,8,9,10,11,12,13,14,15);
        }

        // ---- WMMA burst: 8 MMAs, A frags reused 4x, B frags reused 2x ----
#pragma unroll
        for (int m = 0; m < 2; ++m)
#pragma unroll
            for (int n = 0; n < 4; ++n)
                acc[m][n] = __builtin_amdgcn_wmma_f32_16x16x32_bf16(
                    false, af[m], false, bf[n], (short)0, acc[m][n],
                    false, false);

        // ---- stage tile kb+1 into buffer `nxt` ----
        if (more) {
            bf16x8 a8;
            a8[0] = (__bf16)wv0[0]; a8[1] = (__bf16)wv0[1];
            a8[2] = (__bf16)wv0[2]; a8[3] = (__bf16)wv0[3];
            a8[4] = (__bf16)wv1[0]; a8[5] = (__bf16)wv1[1];
            a8[6] = (__bf16)wv1[2]; a8[7] = (__bf16)wv1[3];
            *(bf16x8*)&As[nxt][ar * 32 + ak] = a8;
#pragma unroll
            for (int i = 0; i < 4; ++i)
                *(bf16x8*)&Bs[nxt][i * 2048 + tid * 8] = bv[i];
        }
        __syncthreads();
    }

    // ---- epilogue: f32 C/D layout — VGPR i: lane<16 -> M=i, lane>=16 -> M=8+i
    OutT* o = Out + (size_t)bc * G * T_DIM;
#pragma unroll
    for (int m = 0; m < 2; ++m) {
#pragma unroll
        for (int n = 0; n < 4; ++n) {
            const int col = tw * 64 + n * 16 + l;
#pragma unroll
            for (int i = 0; i < 8; ++i) {
                const int row = g0 + wg * 32 + m * 16 + h * 8 + i;
                o[(size_t)row * T_DIM + col] = (OutT)acc[m][n][i];
            }
        }
    }
}

// ============================================================================
// Kernel 3: dw1 channel mix (32x8) + squared ReLU, emitting the packed
// WMMA-B layout directly: h2p[(b*32+d)][kb][t][kin].
// ============================================================================
__global__ __launch_bounds__(256) void mix1_sqrelu_pack(
    const __bf16* __restrict__ h1,   // (B, 8, Fh=4096, T) bf16 (unpacked)
    const float* __restrict__ dw1,   // (32, 8)
    __bf16* __restrict__ h2p)        // (B*32, Fh/32=128, T, 32) bf16 packed
{
    constexpr int CIN = 8, COUT = 32, FH = 4096, KB = FH / 32;
    const unsigned i = blockIdx.x * blockDim.x + threadIdx.x; // < 262144
    const int q  = i & 3;                 // kin quarter (8 kin)
    const int t  = (i >> 2) & (T_DIM - 1);
    const int kb = (i >> 10) & (KB - 1);
    const int b  = i >> 17;

    float v[CIN][8];
#pragma unroll
    for (int cc = 0; cc < CIN; ++cc) {
        const __bf16* p =
            h1 + ((size_t)(b * CIN + cc) * FH + kb * 32 + q * 8) * T_DIM + t;
#pragma unroll
        for (int j = 0; j < 8; ++j) v[cc][j] = (float)p[(size_t)j * T_DIM];
    }

    for (int d = 0; d < COUT; ++d) {
        bf16x8 o;
#pragma unroll
        for (int j = 0; j < 8; ++j) {
            float s = 0.f;
#pragma unroll
            for (int cc = 0; cc < CIN; ++cc) s += v[cc][j] * dw1[d * CIN + cc];
            s = fmaxf(s, 0.f);
            o[j] = (__bf16)(s * s);
        }
        __bf16* qp = h2p +
            (((size_t)(b * COUT + d) * KB + kb) * T_DIM + t) * 32 + q * 8;
        *(bf16x8*)qp = o;
    }
}

// ============================================================================
// Kernel 5: dw2 channel mix (8x32) + residual, f32 out.
// ============================================================================
__global__ __launch_bounds__(256) void mix2_residual(
    const float* __restrict__ h3,    // (B, 32, F*T) f32
    const float* __restrict__ dw2,   // (8, 32)
    const float* __restrict__ x,     // (B, 8, F*T)
    float* __restrict__ out)         // (B, 8, F*T)
{
    constexpr int CIN = 32, COUT = 8;
    constexpr size_t FT = (size_t)1024 * T_DIM;
    const size_t i  = (size_t)blockIdx.x * blockDim.x + threadIdx.x;
    const size_t b  = i / FT;
    const size_t ft = i % FT;

    float v[CIN];
    const float* p = h3 + b * CIN * FT + ft;
#pragma unroll
    for (int cc = 0; cc < CIN; ++cc) v[cc] = p[cc * FT];

    for (int d = 0; d < COUT; ++d) {
        float s = 0.f;
#pragma unroll
        for (int cc = 0; cc < CIN; ++cc) s += v[cc] * dw2[d * CIN + cc];
        const size_t o = (b * COUT + d) * FT + ft;
        out[o] = x[o] + s;
    }
}

// ============================================================================
// Launcher.  Inputs: x, ln_w, ln_b, pw1, dw1, pw2, dw2 (all f32).
// Workspace (232 MiB): h0p bf16 8MiB | h1 bf16 32MiB | h2p bf16 128MiB |
//                      h3 f32 64MiB
// ============================================================================
extern "C" void kernel_launch(void* const* d_in, const int* in_sizes, int n_in,
                              void* d_out, int out_size, void* d_ws, size_t ws_size,
                              hipStream_t stream) {
    const float* x    = (const float*)d_in[0];
    const float* ln_w = (const float*)d_in[1];
    const float* ln_b = (const float*)d_in[2];
    const float* pw1  = (const float*)d_in[3];
    const float* dw1  = (const float*)d_in[4];
    const float* pw2  = (const float*)d_in[5];
    const float* dw2  = (const float*)d_in[6];
    float* out = (float*)d_out;

    char* ws = (char*)d_ws;
    __bf16* h0p = (__bf16*)(ws);                        //  8 MiB packed (16,32,256,32)
    __bf16* h1  = (__bf16*)(ws + ((size_t)8   << 20));  // 32 MiB (2,8,4096,256)
    __bf16* h2p = (__bf16*)(ws + ((size_t)40  << 20));  // 128MiB packed (64,128,256,32)
    float*  h3  = (float*) (ws + ((size_t)168 << 20));  // 64 MiB (2,32,1024,256)

    // 1) LayerNorm -> packed bf16
    layernorm_pack_bf16<<<dim3(16), dim3(256), 0, stream>>>(x, ln_w, ln_b, h0p);

    // 2) pw1: G=4096, K=1024, full-T blocks -> h1 (bf16, unpacked)
    gemm_pw_wmma<__bf16><<<dim3(4096 / 64, 1, 2 * 8), dim3(256), 0, stream>>>(
        pw1, h0p, h1, /*Cn=*/8, /*G=*/4096, /*K=*/1024);

    // 3) dw1 mix + squared ReLU -> h2p (bf16, packed)
    mix1_sqrelu_pack<<<dim3(1024), dim3(256), 0, stream>>>(h1, dw1, h2p);

    // 4) pw2: G=1024, K=4096, full-T blocks -> h3 (f32)
    gemm_pw_wmma<float><<<dim3(1024 / 64, 1, 2 * 32), dim3(256), 0, stream>>>(
        pw2, h2p, h3, /*Cn=*/32, /*G=*/1024, /*K=*/4096);

    // 5) dw2 mix + residual -> out (f32)
    mix2_residual<<<dim3(2048), dim3(256), 0, stream>>>(h3, dw2, x, out);
}
